// CosineSimilarityAttention_24154896072767
// MI455X (gfx1250) — matrix operand order repaired and checked by reading, not compiled, not run on hardware
//
#include <hip/hip_runtime.h>
#include <hip/hip_bf16.h>

// CosineSimilarityAttention for MI455X (gfx1250), wave32 + WMMA f16 path.
// Register-blocked WMMA GEMMs: k_scores does 64x32 per wave (8 wmma / 12 b128
// loads per K-step), k_attn_v does 32x32 per wave with LDS-staged B tiles and
// 8-way split-K over batch (f32 partials + reduction).
//
// d_out layout: x [L*H] f32, then attn [B*L*L] f32 (reference returns (x, attn)).
// Workspace layout (~296 MB assumed available):
//   qn_h   [B*L*D] f16      @ 0
//   kn_h   [B*L*D] f16      @ 64 MB
//   keys_h [B*L*D] f16      @ 128 MB
//   attn_h [B*L*L] f16      @ 192 MB
//   out_h  [L*D]   f16      @ 256 MB
//   W_h    [H*D]   f16      @ 258 MB
//   part   [8][L*D] f32     @ 262 MB

typedef _Float16 half_t;
typedef __attribute__((ext_vector_type(16))) _Float16 v16h;
typedef __attribute__((ext_vector_type(8)))  _Float16 h8;
typedef __attribute__((ext_vector_type(8)))  float    v8f;

constexpr int NB = 32;    // batch
constexpr int NL = 1024;  // sequence length
constexpr int ND = 1024;  // feature dim
constexpr int NH = 1024;  // hidden dim
constexpr int SPLIT = 8;  // split-K over batch in k_attn_v (4 batches/split)

// ---- WMMA operand packing (wave32; contiguous k = 16*hi + s order used by this file) -----------------------
// A 16x32 f16: lane m16 = lane&15 holds row M=m16.
//   lanes 0-15 : halves 0-15 = K 0..15
//   lanes 16-31: halves 0-15 = K 16..31
__device__ __forceinline__ v16h load_A_row(const half_t* rowPtr, int hi) {
  const h8 lo = *(const h8*)(rowPtr + 16 * hi);
  const h8 hv = *(const h8*)(rowPtr + 16 * hi + 8);
  v16h a;
#pragma unroll
  for (int i = 0; i < 8; ++i) { a[i] = lo[i]; a[8 + i] = hv[i]; }
  return a;
}

// B 32x16 f16: lane n holds column N = n&15; lanes 0-15 K 0..15, lanes 16-31
// K 16..31 (16 contiguous halves at p).
__device__ __forceinline__ v16h load_B_col16(const half_t* p) {
  const h8 lo = *(const h8*)(p);
  const h8 hv = *(const h8*)(p + 8);
  v16h b;
#pragma unroll
  for (int i = 0; i < 8; ++i) { b[i] = lo[i]; b[8 + i] = hv[i]; }
  return b;
}

__device__ __forceinline__ v8f wmma_f16(v16h a, v16h b, v8f c) {
  return __builtin_amdgcn_wmma_f32_16x16x32_f16(false, a, false, b,
                                                (short)0, c, false, false);
}

// ---- Kernel 1: L2-normalize rows, produce f16 operands ---------------------
// grid: (B*L, 2); y==0 -> query->qn_h, y==1 -> keys->kn_h + raw keys_h
__global__ void k_normalize(const float* __restrict__ q, const float* __restrict__ k,
                            half_t* __restrict__ qn_h, half_t* __restrict__ kn_h,
                            half_t* __restrict__ keys_h) {
  const int row   = blockIdx.x;
  const int which = blockIdx.y;
  const float* src = (which == 0 ? q : k) + (size_t)row * ND;
  __shared__ float red[256];
  float vals[4];
  float ss = 0.f;
#pragma unroll
  for (int i = 0; i < 4; ++i) {
    vals[i] = src[threadIdx.x + 256 * i];
    ss += vals[i] * vals[i];
  }
  red[threadIdx.x] = ss;
  __syncthreads();
  for (int off = 128; off > 0; off >>= 1) {
    if ((int)threadIdx.x < off) red[threadIdx.x] += red[threadIdx.x + off];
    __syncthreads();
  }
  const float rn = rsqrtf(red[0]);
  half_t* dstN = (which == 0 ? qn_h : kn_h) + (size_t)row * ND;
#pragma unroll
  for (int i = 0; i < 4; ++i) dstN[threadIdx.x + 256 * i] = (half_t)(vals[i] * rn);
  if (which == 1) {
    half_t* dk = keys_h + (size_t)row * ND;
#pragma unroll
    for (int i = 0; i < 4; ++i) dk[threadIdx.x + 256 * i] = (half_t)vals[i];
  }
}

// ---- Kernel 2: scores[b,i,j] = <kn[b,i], qn[b,j]> via WMMA -----------------
// Register-blocked: each wave computes a 64x32 region = 4 M-tiles x 2 N-tiles.
// Per K-step: 8 b128 A loads + 4 b128 B loads feed 8 wmma.
// jobs = B * (L/64) * (L/32) = 16384; 8 waves/block -> 2048 blocks.
__global__ void k_scores(const half_t* __restrict__ kn_h, const half_t* __restrict__ qn_h,
                         float* __restrict__ scores) {
  const int lane = threadIdx.x & 31;
  const int wave = threadIdx.x >> 5;
  const int job  = blockIdx.x * 8 + wave;         // 0 .. NB*512-1
  const int b    = job >> 9;
  const int r    = job & 511;
  const int it   = (r >> 5) << 6;                 // 64-row tile (key rows, M)
  const int jt   = (r & 31) << 5;                 // 32-col tile (query cols, N)
  const int m16  = lane & 15, hi = lane >> 4;

  const half_t* aBase = kn_h + ((size_t)b * NL + (it + m16)) * ND;
  const half_t* bBase = qn_h + ((size_t)b * NL + (jt + m16)) * ND + 16 * hi;

  v8f c[4][2] = {};
  for (int d0 = 0; d0 < ND; d0 += 32) {
    __builtin_prefetch(aBase + d0 + 64, 0, 0);    // global_prefetch_b8
    __builtin_prefetch(bBase + d0 + 64, 0, 0);
    v16h a[4], bm[2];
#pragma unroll
    for (int mi = 0; mi < 4; ++mi)
      a[mi] = load_A_row(aBase + (size_t)(16 * mi) * ND + d0, hi);
#pragma unroll
    for (int ni = 0; ni < 2; ++ni)
      bm[ni] = load_B_col16(bBase + (size_t)(16 * ni) * ND + d0);
#pragma unroll
    for (int mi = 0; mi < 4; ++mi)
#pragma unroll
      for (int ni = 0; ni < 2; ++ni)
        c[mi][ni] = wmma_f16(a[mi], bm[ni], c[mi][ni]);
  }
  // C/D layout: VGPR rr -> M = 16*mi + 8*hi + rr, N = 16*ni + m16
  float* ob = scores + (size_t)b * NL * NL;
#pragma unroll
  for (int mi = 0; mi < 4; ++mi) {
#pragma unroll
    for (int ni = 0; ni < 2; ++ni) {
      float* o = ob + (size_t)(it + 16 * mi + 8 * hi) * NL + (jt + 16 * ni + m16);
#pragma unroll
      for (int rr = 0; rr < 8; ++rr) o[(size_t)rr * NL] = c[mi][ni][rr];
    }
  }
}

// ---- Kernel 3: row softmax in place + f16 copy -----------------------------
// One block (256 threads) per row of length L. grid.x = B*L.
__global__ void k_softmax(float* __restrict__ attn, half_t* __restrict__ attn_h) {
  const size_t row = blockIdx.x;
  float* p = attn + row * (size_t)NL;
  __shared__ float red[256];
  float v[4];
  float mx = -3.0e38f;
#pragma unroll
  for (int i = 0; i < 4; ++i) {
    v[i] = p[threadIdx.x + 256 * i];
    mx = fmaxf(mx, v[i]);
  }
  red[threadIdx.x] = mx;
  __syncthreads();
  for (int off = 128; off > 0; off >>= 1) {
    if ((int)threadIdx.x < off) red[threadIdx.x] = fmaxf(red[threadIdx.x], red[threadIdx.x + off]);
    __syncthreads();
  }
  mx = red[0];
  __syncthreads();
  float s = 0.f;
#pragma unroll
  for (int i = 0; i < 4; ++i) {
    v[i] = __expf(v[i] - mx);
    s += v[i];
  }
  red[threadIdx.x] = s;
  __syncthreads();
  for (int off = 128; off > 0; off >>= 1) {
    if ((int)threadIdx.x < off) red[threadIdx.x] += red[threadIdx.x + off];
    __syncthreads();
  }
  const float inv = 1.0f / red[0];
  half_t* ph = attn_h + row * (size_t)NL;
#pragma unroll
  for (int i = 0; i < 4; ++i) {
    const float a = v[i] * inv;
    p[threadIdx.x + 256 * i]  = a;
    ph[threadIdx.x + 256 * i] = (half_t)a;
  }
}

// ---- Kernel 4: partial out[i,d] = sum_{b in split} attn[b,i,:]@keys[b,:,d] -
// Each wave: 32x32 region (2x2 tiles). Block: 8 waves = 256 rows x 32 cols,
// keys B-tile (32x32) staged transposed through LDS, shared by all 8 waves.
// grid: (L/256, D/32, SPLIT). Partials are f32, reduced by k_reduce_out.
__global__ void k_attn_v(const half_t* __restrict__ attn_h, const half_t* __restrict__ keys_h,
                         float* __restrict__ part) {
  const int lane = threadIdx.x & 31;
  const int wave = threadIdx.x >> 5;
  const int it   = blockIdx.x * 256 + wave * 32;
  const int dt   = blockIdx.y * 32;
  const int sp   = blockIdx.z;
  const int m16  = lane & 15, hi = lane >> 4;

  __shared__ __align__(32) half_t ldsB[32 * 32];  // [col 0..31][k 0..31]

  v8f c[2][2] = {};
  for (int bb = 0; bb < NB / SPLIT; ++bb) {
    const int b = sp * (NB / SPLIT) + bb;
    const half_t* aBase = attn_h + ((size_t)b * NL + (it + m16)) * (size_t)NL;
    const size_t  kbase = (size_t)b * NL * ND;
    for (int j0 = 0; j0 < NL; j0 += 32) {
      __syncthreads();
      // stage keys[b, j0+k, dt+c] transposed: ldsB[c*32+k]; coalesced reads.
      for (int e = threadIdx.x; e < 1024; e += 256) {
        const int kk = e >> 5, cc = e & 31;
        ldsB[cc * 32 + kk] = keys_h[kbase + (size_t)(j0 + kk) * ND + dt + cc];
      }
      __syncthreads();
      v16h a0 = load_A_row(aBase + j0, hi);
      v16h a1 = load_A_row(aBase + (size_t)16 * NL + j0, hi);
      v16h b0 = load_B_col16(&ldsB[m16 * 32 + 16 * hi]);
      v16h b1 = load_B_col16(&ldsB[(16 + m16) * 32 + 16 * hi]);
      c[0][0] = wmma_f16(a0, b0, c[0][0]);
      c[0][1] = wmma_f16(a0, b1, c[0][1]);
      c[1][0] = wmma_f16(a1, b0, c[1][0]);
      c[1][1] = wmma_f16(a1, b1, c[1][1]);
    }
  }
  float* ob = part + (size_t)sp * NL * ND;
#pragma unroll
  for (int mi = 0; mi < 2; ++mi) {
#pragma unroll
    for (int ni = 0; ni < 2; ++ni) {
      float* o = ob + (size_t)(it + 16 * mi + 8 * hi) * ND + (dt + 16 * ni + m16);
#pragma unroll
      for (int rr = 0; rr < 8; ++rr) o[(size_t)rr * ND] = c[mi][ni][rr];
    }
  }
}

// ---- Kernel 5: reduce split-K partials -> out_h (f16) ----------------------
__global__ void k_reduce_out(const float* __restrict__ part, half_t* __restrict__ out_h) {
  const int i = blockIdx.x * 256 + threadIdx.x;   // 0 .. L*D-1
  float s = 0.f;
#pragma unroll
  for (int sp = 0; sp < SPLIT; ++sp) s += part[(size_t)sp * NL * ND + i];
  out_h[i] = (half_t)s;
}

// ---- Kernel 6: W f32 -> f16 ------------------------------------------------
__global__ void k_cvt_w(const float* __restrict__ W, half_t* __restrict__ W_h, int n) {
  const int i = blockIdx.x * 256 + threadIdx.x;
  if (i < n) W_h[i] = (half_t)W[i];
}

// ---- Kernel 7: x = relu(out @ W^T + bias) via WMMA -------------------------
// One 16x16 tile per wave; grid.x = 64*64/8 (small GEMM, parallelism first).
__global__ void k_fc_relu(const half_t* __restrict__ out_h, const half_t* __restrict__ W_h,
                          const float* __restrict__ bias, float* __restrict__ x) {
  const int lane = threadIdx.x & 31;
  const int wave = threadIdx.x >> 5;
  const int tile = blockIdx.x * 8 + wave;         // 0..4095
  const int lt   = (tile >> 6) << 4;              // row tile (l)
  const int ht   = (tile & 63) << 4;              // col tile (h)
  const int m16  = lane & 15, hi = lane >> 4;

  const half_t* aRow = out_h + (size_t)(lt + m16) * ND;
  const half_t* bRow = W_h + (size_t)(ht + m16) * ND + 16 * hi;  // B[k=d][n=h] = W[h,d]

  v8f c = {};
  for (int d0 = 0; d0 < ND; d0 += 32) {
    __builtin_prefetch(aRow + d0 + 64, 0, 0);
    __builtin_prefetch(bRow + d0 + 64, 0, 0);
    v16h a  = load_A_row(aRow + d0, hi);
    v16h bm = load_B_col16(bRow + d0);
    c = wmma_f16(a, bm, c);
  }
  const float bb = bias[ht + m16];
  float* o = x + (size_t)(lt + 8 * hi) * NH + ht + m16;
#pragma unroll
  for (int rr = 0; rr < 8; ++rr) o[(size_t)rr * NH] = fmaxf(c[rr] + bb, 0.0f);
}

// ---- Launch ----------------------------------------------------------------
extern "C" void kernel_launch(void* const* d_in, const int* in_sizes, int n_in,
                              void* d_out, int out_size, void* d_ws, size_t ws_size,
                              hipStream_t stream) {
  const float* q    = (const float*)d_in[0];  // [B,L,D]
  const float* k    = (const float*)d_in[1];  // [B,L,D]
  const float* W    = (const float*)d_in[2];  // [H,D]
  const float* bias = (const float*)d_in[3];  // [H]

  float* x    = (float*)d_out;                       // [L,H]
  float* attn = (float*)d_out + (size_t)NL * NH;     // [B,L,L]

  char* ws = (char*)d_ws;
  const size_t SZ_BLD = (size_t)NB * NL * ND * sizeof(half_t);  // 64 MB
  half_t* qn_h   = (half_t*)(ws);
  half_t* kn_h   = (half_t*)(ws + SZ_BLD);
  half_t* keys_h = (half_t*)(ws + 2 * SZ_BLD);
  half_t* attn_h = (half_t*)(ws + 3 * SZ_BLD);      // B*L*L f16 = 64 MB
  half_t* out_h  = (half_t*)(ws + 4 * SZ_BLD);      // L*D f16 = 2 MB
  half_t* W_h    = (half_t*)(ws + 4 * SZ_BLD + (size_t)NL * ND * sizeof(half_t));
  float*  part   = (float*)(ws + 4 * SZ_BLD + 2 * (size_t)NL * ND * sizeof(half_t)
                               + (size_t)NH * ND * sizeof(half_t)); // 32 MB

  // 1. normalize + f16 conversion
  k_normalize<<<dim3(NB * NL, 2), 256, 0, stream>>>(q, k, qn_h, kn_h, keys_h);
  // 2. scores (into d_out attn region), 64x32 per wave
  k_scores<<<dim3(NB * 512 / 8), 256, 0, stream>>>(kn_h, qn_h, attn);
  // 3. softmax in place + f16 copy
  k_softmax<<<dim3(NB * NL), 256, 0, stream>>>(attn, attn_h);
  // 4. split-K partials of out = sum_b attn @ keys
  k_attn_v<<<dim3(NL / 256, ND / 32, SPLIT), 256, 0, stream>>>(attn_h, keys_h, part);
  // 5. reduce partials -> out_h f16
  k_reduce_out<<<dim3(NL * ND / 256), 256, 0, stream>>>(part, out_h);
  // 6. W -> f16
  k_cvt_w<<<dim3((NH * ND + 255) / 256), 256, 0, stream>>>(W, W_h, NH * ND);
  // 7. x = relu(out @ W^T + b)
  k_fc_relu<<<dim3(64 * 64 / 8), 256, 0, stream>>>(out_h, W_h, bias, x);
}
